// VanillaLSTMNet_19207093747713
// MI455X (gfx1250) — matrix operand order repaired and checked by reading, not compiled, not run on hardware
//
#include <hip/hip_runtime.h>
#include <hip/hip_bf16.h>

// ---------------- problem constants (from reference) ----------------
#define IN_DIM   2
#define EMB      64
#define RNN      128
#define GATES    512           // 4*RNN, order i,f,g,o
#define SEQ      20
#define PRED     30
#define TOTAL_T  (SEQ + PRED)  // 50
#define PEDS     16384

// tiling: 4 wave32 per WG, each wave owns 32 peds (two 16-row M tiles)
#define PEDS_PER_WG 128
#define THREADS     128
#define M_PER_WAVE  2
#define NT_GATES    32         // 512 / 16
#define KT_E        2          // 64  / 32
#define KT_H        4          // 128 / 32
#define KT_ALL      6          // e (2) + h (4) k-tiles per gate column

typedef _Float16 v16h __attribute__((ext_vector_type(16)));
typedef _Float16 v8h  __attribute__((ext_vector_type(8)));
typedef float    v8f  __attribute__((ext_vector_type(8)));

// ---------------- LDS layout (dynamic shared) ----------------
#define WIH_HALVES (KT_E * NT_GATES * 32 * 16)   // 32768  (64 KB)
#define WHH_HALVES (KT_H * NT_GATES * 32 * 16)   // 65536  (128 KB)
#define H_HALVES   (PEDS_PER_WG * RNN)           // 16384  (32 KB)
#define E_HALVES   (PEDS_PER_WG * EMB)           // 8192   (16 KB)
#define C_FLOATS   (PEDS_PER_WG * RNN)           // 16384  (64 KB)
#define SMEM_BYTES ((WIH_HALVES + WHH_HALVES + H_HALVES + E_HALVES) * 2 + \
                    (C_FLOATS + 256 + 512 + 128 + 64 + 256 + 2) * 4) // 316168 B

__device__ __forceinline__ v8f wmma16(v16h a, v16h b, v8f c) {
  return __builtin_amdgcn_wmma_f32_16x16x32_f16(false, a, false, b, (short)0, c,
                                                false, false);
}

// A-operand: lane (hi = lane>>4, m = lane&15) holds row m, K chunk
// [kbase .. kbase+15] (contiguous, matching the B packing)
__device__ __forceinline__ v16h load_a(const _Float16* buf, int row,
                                       int rowStride, int kbase) {
  const v8h lo = *(const v8h*)(buf + row * rowStride + kbase);
  const v8h hi = *(const v8h*)(buf + row * rowStride + kbase + 8);
  return __builtin_shufflevector(lo, hi, 0, 1, 2, 3, 4, 5, 6, 7, 8, 9, 10, 11,
                                 12, 13, 14, 15);
}

__device__ __forceinline__ float sigf(float x) {
  return 1.0f / (1.0f + __expf(-x));
}
__device__ __forceinline__ float tanhf_fast(float x) {
  float e = __expf(2.0f * x);
  return 1.0f - 2.0f / (e + 1.0f);   // safe at +/- inf
}

__global__ __launch_bounds__(THREADS)
void lstm_traj_kernel(const float* __restrict__ obs,    // [SEQ][PEDS][2]
                      const float* __restrict__ Wemb,   // [EMB][2]
                      const float* __restrict__ Bemb,   // [EMB]
                      const float* __restrict__ Wih,    // [512][64]
                      const float* __restrict__ Bih,    // [512]
                      const float* __restrict__ Whh,    // [512][128]
                      const float* __restrict__ Bhh,    // [512]
                      const float* __restrict__ Wout,   // [2][128]
                      const float* __restrict__ Bout,   // [2]
                      float* __restrict__ out)          // [PRED][PEDS][2]
{
  extern __shared__ __align__(32) char smem_raw[];
  _Float16* sWih = (_Float16*)smem_raw;
  _Float16* sWhh = sWih + WIH_HALVES;
  _Float16* sH   = sWhh + WHH_HALVES;
  _Float16* sE   = sH + H_HALVES;
  float* sC    = (float*)(sE + E_HALVES);  // cell state, fragment layout
  float* sOut  = sC + C_FLOATS;            // [128][2]
  float* sBias = sOut + 256;               // [512] = b_ih + b_hh
  float* sWemb = sBias + 512;              // [64][2]
  float* sBemb = sWemb + 128;              // [64]
  float* sWout = sBemb + 64;               // [2][128]
  float* sBout = sWout + 256;              // [2]

  const int tid  = threadIdx.x;
  const int lane = tid & 31;
  const int wv   = tid >> 5;        // wave 0..3, owns 32 ped rows
  const int m    = lane & 15;       // row / column within 16-tile
  const int hi   = lane >> 4;       // lane-half select
  const int pedBase = blockIdx.x * PEDS_PER_WG;

  // ---- one-time: pack weights into B-fragment order (fp16), load params ----
  // B[k][n] = W[n][k]; fragment element j of lane l: k = kt*32 + 16*(l>>4) + j
  for (int u = tid; u < KT_E * NT_GATES * 32; u += THREADS) {
    int l = u & 31, fi = u >> 5;
    int kt = fi >> 5, nt = fi & 31;
    int n = nt * 16 + (l & 15);
    int kb = kt * 32 + 16 * (l >> 4);
#pragma unroll
    for (int j = 0; j < 16; ++j)
      sWih[u * 16 + j] = (_Float16)Wih[n * EMB + kb + j];
  }
  for (int u = tid; u < KT_H * NT_GATES * 32; u += THREADS) {
    int l = u & 31, fi = u >> 5;
    int kt = fi >> 5, nt = fi & 31;
    int n = nt * 16 + (l & 15);
    int kb = kt * 32 + 16 * (l >> 4);
#pragma unroll
    for (int j = 0; j < 16; ++j)
      sWhh[u * 16 + j] = (_Float16)Whh[n * RNN + kb + j];
  }
  for (int i = tid; i < H_HALVES; i += THREADS) sH[i] = (_Float16)0.0f;
  for (int i = tid; i < C_FLOATS; i += THREADS) sC[i] = 0.0f;
  for (int i = tid; i < 512; i += THREADS) sBias[i] = Bih[i] + Bhh[i];
  for (int i = tid; i < 128; i += THREADS) sWemb[i] = Wemb[i];
  for (int i = tid; i < 64; i += THREADS) sBemb[i] = Bemb[i];
  for (int i = tid; i < 256; i += THREADS) sWout[i] = Wout[i];
  for (int i = tid; i < 2; i += THREADS) sBout[i] = Bout[i];
  __syncthreads();

  // ---- per-lane base pointers: B fragments become base + const-offset ----
  const _Float16* pWihK[KT_E];
#pragma unroll
  for (int ke = 0; ke < KT_E; ++ke)
    pWihK[ke] = sWih + (ke * NT_GATES * 32 + lane) * 16;
  const _Float16* pWhhK[KT_H];
#pragma unroll
  for (int kh = 0; kh < KT_H; ++kh)
    pWhhK[kh] = sWhh + (kh * NT_GATES * 32 + lane) * 16;
  // c-state fragment pointer: tile (mt, nb) at const offset (mt*8+nb)*1024 B
  float* pC = sC + wv * (M_PER_WAVE * 8 * 32 * 8) + lane * 8;

  // unified B loader: q = 0..1 -> W_ih ktile, q = 2..5 -> W_hh ktile
  auto loadB = [&](int q, int nt) -> v16h {
    if (q < KT_E) return *(const v16h*)(pWihK[q] + nt * 512);
    return *(const v16h*)(pWhhK[q - KT_E] + nt * 512);
  };

  for (int t = 0; t < TOTAL_T; ++t) {
    // ---- embedding: e = relu(x @ Wembᵀ + b) ; x = obs (encode) | out (decode)
    for (int u = lane; u < 32 * EMB; u += 32) {
      int lm = u >> 6, j = u & 63;           // lm 0..31, j 0..63
      int lp = wv * 32 + lm;
      float x0, x1;
      if (t < SEQ) {
        const float* xp = obs + ((size_t)t * PEDS + pedBase + lp) * 2;
        x0 = xp[0]; x1 = xp[1];
      } else {
        x0 = sOut[lp * 2 + 0]; x1 = sOut[lp * 2 + 1];
      }
      float v = fmaf(x0, sWemb[j * 2 + 0], fmaf(x1, sWemb[j * 2 + 1], sBemb[j]));
      sE[lp * EMB + j] = (_Float16)(v > 0.0f ? v : 0.0f);
    }
    __syncthreads();

    // ---- unified A fragments: [mt][0..1] = e ktiles, [mt][2..5] = h ktiles
    v16h aX[M_PER_WAVE][KT_ALL];
#pragma unroll
    for (int mt = 0; mt < M_PER_WAVE; ++mt) {
      const int rowL = wv * 32 + mt * 16 + m;
#pragma unroll
      for (int ke = 0; ke < KT_E; ++ke)
        aX[mt][ke] = load_a(sE, rowL, EMB, ke * 32 + 16 * hi);
#pragma unroll
      for (int kh = 0; kh < KT_H; ++kh)
        aX[mt][KT_E + kh] = load_a(sH, rowL, RNN, kh * 32 + 16 * hi);
    }

    // ---- 32 stages (nb 0..7 x gate 0..3), explicit B double-buffering ----
    v16h bufs[2][KT_ALL];
    v8f  acc[M_PER_WAVE][4];
    v8f  cOldv[M_PER_WAVE];
    float bI = 0.f, bF = 0.f, bG = 0.f, bO = 0.f;
    const int ncolBase = m;
#pragma unroll
    for (int q = 0; q < KT_ALL; ++q)        // prefetch stage 0 (nb=0, g=0 -> nt=0)
      bufs[0][q] = loadB(q, 0);
#pragma unroll
    for (int s = 0; s < 32; ++s) {
      const int nb = s >> 2, g = s & 3;
      const int cur = s & 1, nxt = cur ^ 1;
      // prefetch next stage's 6 B fragments into the other buffer half
      if (s < 31) {
        const int s2 = s + 1;
        const int nt2 = (s2 & 3) * 8 + (s2 >> 2);
#pragma unroll
        for (int q = 0; q < KT_ALL; ++q) bufs[nxt][q] = loadB(q, nt2);
      }
      if (g == 0) {   // early loads for this nb's epilogue (used 3 stages later)
        const int ncol = nb * 16 + ncolBase;
        bI = sBias[0 * RNN + ncol];
        bF = sBias[1 * RNN + ncol];
        bG = sBias[2 * RNN + ncol];
        bO = sBias[3 * RNN + ncol];
#pragma unroll
        for (int mt = 0; mt < M_PER_WAVE; ++mt)
          cOldv[mt] = *(const v8f*)(pC + (mt * 8 + nb) * 256);
      }
      // 12 WMMAs for this (nb, gate): one B fragment feeds both M tiles
      acc[0][g] = v8f{};
      acc[1][g] = v8f{};
#pragma unroll
      for (int q = 0; q < KT_ALL; ++q) {
        acc[0][g] = wmma16(aX[0][q], bufs[cur][q], acc[0][g]);
        acc[1][g] = wmma16(aX[1][q], bufs[cur][q], acc[1][g]);
      }
      // epilogue for this nb once all 4 gates are accumulated
      if (g == 3) {
#pragma unroll
        for (int mt = 0; mt < M_PER_WAVE; ++mt) {
          v8f cNew;
#pragma unroll
          for (int v = 0; v < 8; ++v) {
            float iv = sigf(acc[mt][0][v] + bI);
            float fv = sigf(acc[mt][1][v] + bF);
            float gv = tanhf_fast(acc[mt][2][v] + bG);
            float ov = sigf(acc[mt][3][v] + bO);
            float cn = fmaf(fv, cOldv[mt][v], iv * gv);
            cNew[v] = cn;
            float hn = ov * tanhf_fast(cn);
            // C-layout -> sH[ped][k]: M = v + 8*hi, N = m
            sH[(wv * 32 + mt * 16 + v + 8 * hi) * RNN + nb * 16 + m] =
                (_Float16)hn;
          }
          *(v8f*)(pC + (mt * 8 + nb) * 256) = cNew;
        }
      }
    }

    // ---- output head: out = h @ Woutᵀ + b  (needed from t=19 on)
    if (t >= SEQ - 1) {
      for (int u = lane; u < 32 * 2; u += 32) {   // 32 peds x 2 outputs
        const int p = u >> 1, o = u & 1;
        const int lp = wv * 32 + p;
        float acc2 = sBout[o];
#pragma unroll 8
        for (int k = 0; k < RNN; ++k)
          acc2 = fmaf((float)sH[lp * RNN + k], sWout[o * RNN + k], acc2);
        sOut[lp * 2 + o] = acc2;
        if (t >= SEQ)
          out[(((size_t)(t - SEQ)) * PEDS + pedBase + lp) * 2 + o] = acc2;
      }
    }
    __syncthreads();
  }
}

extern "C" void kernel_launch(void* const* d_in, const int* in_sizes, int n_in,
                              void* d_out, int out_size, void* d_ws,
                              size_t ws_size, hipStream_t stream) {
  const float* obs  = (const float*)d_in[0];
  const float* Wemb = (const float*)d_in[1];
  const float* Bemb = (const float*)d_in[2];
  const float* Wih  = (const float*)d_in[3];
  const float* Bih  = (const float*)d_in[4];
  const float* Whh  = (const float*)d_in[5];
  const float* Bhh  = (const float*)d_in[6];
  const float* Wout = (const float*)d_in[7];
  const float* Bout = (const float*)d_in[8];
  float* out = (float*)d_out;

  dim3 grid(PEDS / PEDS_PER_WG);   // 128 workgroups -> 128 WGPs
  dim3 block(THREADS);             // 4 wave32 each (1 per SIMD32)
  lstm_traj_kernel<<<grid, block, SMEM_BYTES, stream>>>(
      obs, Wemb, Bemb, Wih, Bih, Whh, Bhh, Wout, Bout, out);
}
